// MultiHeadAttention_36086315221183
// MI455X (gfx1250) — compile-verified, matrix operand order checked by reading, not run on hardware
//
#include <hip/hip_runtime.h>
#include <hip/hip_bf16.h>
#include <cstdint>
#include <cstddef>

#define D_MODEL   1024
#define NUM_HEADS 16
#define HEAD_DIM  64
#define SEQ       2048
#define BATCH     2
#define NTOK      (BATCH * SEQ)   // 4096 tokens

typedef __attribute__((ext_vector_type(16))) __bf16       v16bf;
typedef __attribute__((ext_vector_type(8)))  float        v8f;
typedef __attribute__((ext_vector_type(4)))  unsigned int u32x4;

union FragA { v16bf v; u32x4 q[2]; };
union FragC { v8f   v; float f[8]; };

// round-to-nearest-even f32 -> bf16
__device__ __forceinline__ unsigned short f2bf(float x) {
  union { float f; unsigned u; } c; c.f = x;
  unsigned r = c.u + 0x7FFFu + ((c.u >> 16) & 1u);
  return (unsigned short)(r >> 16);
}

// Load a 16x32 bf16 A-fragment (or the mirrored 32x16 B-fragment) from a
// row-major [rows x ld] bf16 matrix, per the CDNA5 16-bit WMMA layout:
//   lane<16 : row = row0+lane,    K = k0+{0..7, 16..23}
//   lane>=16: row = row0+lane-16, K = k0+{8..15, 24..31}
// Two 16-byte loads per lane -> global_load_b128 / ds_load_b128.
__device__ __forceinline__ v16bf load_frag(const unsigned short* base, int row0,
                                           int ld, int k0, int lane) {
  const int half = lane >> 4;
  const int r    = row0 + (lane & 15);
  const unsigned short* p = base + (size_t)r * (size_t)ld + k0 + half * 8;
  FragA f;
  f.q[0] = *(const u32x4*)(p);
  f.q[1] = *(const u32x4*)(p + 16);
  return f.v;
}

__device__ __forceinline__ v8f wmma_bf16(v16bf a, v16bf b, v8f c) {
  // D = A(16x32) * B(32x16) + C, f32 accumulate
  return __builtin_amdgcn_wmma_f32_16x16x32_bf16(
      /*neg_a=*/false, a, /*neg_b=*/false, b,
      /*c_mod=*/(short)0, c, /*reuse_a=*/false, /*reuse_b=*/false);
}

// Issue one 16-byte async global->LDS copy (ASYNCcnt-tracked).
__device__ __forceinline__ void async_b128(const unsigned short* gsrc,
                                           unsigned short* lds_dst) {
  unsigned ldst = (unsigned)(uintptr_t)lds_dst;   // LDS byte offset (low 32 bits)
  unsigned long long gaddr = (unsigned long long)(uintptr_t)gsrc;
  asm volatile("global_load_async_to_lds_b128 %0, %1, off"
               :: "v"(ldst), "v"(gaddr) : "memory");
}

__device__ __forceinline__ void wait_async0() {
  asm volatile("s_wait_asynccnt 0" ::: "memory");
}

// ---------------------------------------------------------------- conversion
__global__ void __launch_bounds__(256)
cvt_kernel(const float* __restrict__ src, unsigned short* __restrict__ dst, int n) {
  int i = blockIdx.x * blockDim.x + threadIdx.x;
  if (i < n) dst[i] = f2bf(src[i]);
}

// ---------------------------------------------------------------- QKV GEMM
// out[t,n] = X[t,:] . Wqkv[n,:] + bqkv[n]; route n into per-head Q/K/Vt (bf16).
// Block = 8 waves sharing one 64-column weight strip; the 64x32 bf16 weight
// tile is staged into LDS with async copies (double-buffered), so weight
// cache traffic drops 8x. Each wave owns a 16x64 output strip.
__global__ void __launch_bounds__(256)
qkv_gemm_kernel(const unsigned short* __restrict__ Xb,
                const unsigned short* __restrict__ Wb,
                const float* __restrict__ bias,
                unsigned short* __restrict__ Q,
                unsigned short* __restrict__ K,
                unsigned short* __restrict__ Vt) {
  __shared__ unsigned short bbuf[2][64 * 32];   // 4 KB per buffer

  const int tid  = threadIdx.x;
  const int lane = tid & 31;
  const int warp = tid >> 5;
  const int nt   = blockIdx.x % 48;            // 64-col weight strip
  const int mt   = (blockIdx.x / 48) * 8 + warp;
  const int n0   = nt * 64;

  // staging assignment: 256 threads x 16B = 4 KB tile
  const int brow = tid >> 2;                   // 0..63
  const int bcol = (tid & 3) * 8;              // 0,8,16,24

  const v8f zro = {0.f, 0.f, 0.f, 0.f, 0.f, 0.f, 0.f, 0.f};
  FragC acc[4];
#pragma unroll
  for (int j = 0; j < 4; ++j) acc[j].v = zro;

  // prologue: stage k0 = 0
  async_b128(Wb + (size_t)(n0 + brow) * D_MODEL + bcol, &bbuf[0][brow * 32 + bcol]);

  int cur = 0;
  for (int k0 = 0; k0 < D_MODEL; k0 += 32) {
    wait_async0();
    __syncthreads();                            // bbuf[cur] ready for all waves
    if (k0 + 32 < D_MODEL)
      async_b128(Wb + (size_t)(n0 + brow) * D_MODEL + (k0 + 32) + bcol,
                 &bbuf[cur ^ 1][brow * 32 + bcol]);

    // hoist all fragment loads so one wait covers them and the 4 WMMAs
    // issue back-to-back (WMMA co-executes with trailing VALU/LDS work)
    v16bf a  = load_frag(Xb, mt * 16, D_MODEL, k0, lane);
    v16bf b0 = load_frag(bbuf[cur],  0, 32, 0, lane);
    v16bf b1 = load_frag(bbuf[cur], 16, 32, 0, lane);
    v16bf b2 = load_frag(bbuf[cur], 32, 32, 0, lane);
    v16bf b3 = load_frag(bbuf[cur], 48, 32, 0, lane);
    acc[0].v = wmma_bf16(a, b0, acc[0].v);
    acc[1].v = wmma_bf16(a, b1, acc[1].v);
    acc[2].v = wmma_bf16(a, b2, acc[2].v);
    acc[3].v = wmma_bf16(a, b3, acc[3].v);
    cur ^= 1;
  }

  const int half = lane >> 4;
#pragma unroll
  for (int j = 0; j < 4; ++j) {
    const int n = n0 + j * 16 + (lane & 15);
    const int h = n / 192;
    const int r = n - h * 192;
    const float bv = bias[n];
#pragma unroll
    for (int i = 0; i < 8; ++i) {
      const int t = mt * 16 + i + 8 * half;     // token
      const int b = t >> 11, s = t & (SEQ - 1);
      const unsigned short val = f2bf(acc[j].f[i] + bv);
      const size_t bh = (size_t)(b * NUM_HEADS + h);
      if (r < 64)
        Q[(bh * SEQ + s) * HEAD_DIM + r] = val;
      else if (r < 128)
        K[(bh * SEQ + s) * HEAD_DIM + (r - 64)] = val;
      else
        Vt[(bh * HEAD_DIM + (r - 128)) * SEQ + s] = val;   // V stored transposed
    }
  }
}

// ---------------------------------------------------------------- attention
// Block = 8 waves on one (b,h), covering 128 queries (8 q-tiles). The 32-key
// K tile (32x64) and V tile (64x32, from transposed V) are staged into LDS
// once per block with async copies (double-buffered) and shared by all waves:
// 8x less K/V cache traffic. Flash-style online softmax; P goes C-layout ->
// LDS -> A-layout within each wave.
__global__ void __launch_bounds__(256)
attn_kernel(const unsigned short* __restrict__ Q,
            const unsigned short* __restrict__ K,
            const unsigned short* __restrict__ Vt,
            unsigned short* __restrict__ Ctx) {
  __shared__ unsigned short kbuf[2][32 * 64];   // keys x hd, 4 KB each
  __shared__ unsigned short vbuf[2][64 * 32];   // hd x keys, 4 KB each
  __shared__ unsigned short plds[8][512];       // per-wave 16x32 bf16 P tile

  const int tid  = threadIdx.x;
  const int lane = tid & 31;
  const int warp = tid >> 5;
  const int bh   = blockIdx.x >> 4;             // b*16 + h
  const int qt   = (blockIdx.x & 15) * 8 + warp;
  const int half = lane >> 4;

  const unsigned short* Qb = Q  + (size_t)bh * SEQ * HEAD_DIM;
  const unsigned short* Kb = K  + (size_t)bh * SEQ * HEAD_DIM;
  const unsigned short* Vb = Vt + (size_t)bh * HEAD_DIM * SEQ;
  unsigned short* pl = plds[warp];

  // staging assignments: 256 threads x 16B per tile
  const int krow = tid >> 3, kcol = (tid & 7) * 8;   // K tile [32][64]
  const int vrow = tid >> 2, vcol = (tid & 3) * 8;   // V tile [64][32]

  // Q tile resident in registers: A-fragments for hd k-steps 0..31, 32..63
  const v16bf qa0 = load_frag(Qb, qt * 16, HEAD_DIM, 0,  lane);
  const v16bf qa1 = load_frag(Qb, qt * 16, HEAD_DIM, 32, lane);

  const v8f zro = {0.f, 0.f, 0.f, 0.f, 0.f, 0.f, 0.f, 0.f};
  FragC o[4];
#pragma unroll
  for (int j = 0; j < 4; ++j) o[j].v = zro;
  float mrow[8], lrow[8];
#pragma unroll
  for (int i = 0; i < 8; ++i) { mrow[i] = -1e30f; lrow[i] = 0.f; }

  const float scale = 0.125f;   // 1/sqrt(64)

  // prologue: stage key tile 0
  async_b128(Kb + (size_t)krow * HEAD_DIM + kcol, &kbuf[0][krow * 64 + kcol]);
  async_b128(Vb + (size_t)vrow * SEQ + vcol,      &vbuf[0][vrow * 32 + vcol]);

  int cur = 0;
  for (int kt = 0; kt < SEQ / 32; ++kt) {
    wait_async0();
    __syncthreads();                            // K/V tiles ready for all waves
    if (kt + 1 < SEQ / 32) {
      const int nb = (kt + 1) * 32;
      async_b128(Kb + (size_t)(nb + krow) * HEAD_DIM + kcol,
                 &kbuf[cur ^ 1][krow * 64 + kcol]);
      async_b128(Vb + (size_t)vrow * SEQ + nb + vcol,
                 &vbuf[cur ^ 1][vrow * 32 + vcol]);
    }

    const unsigned short* Kc = kbuf[cur];
    const unsigned short* Vc = vbuf[cur];

    // scores: two 16x16 tiles (keys +0..15 and +16..31); hoist all K-frag
    // loads so the 4 WMMAs issue back-to-back after one wait
    FragC s0, s1;
    {
      v16bf b00 = load_frag(Kc, 0,  64, 0,  lane);
      v16bf b01 = load_frag(Kc, 0,  64, 32, lane);
      v16bf b10 = load_frag(Kc, 16, 64, 0,  lane);
      v16bf b11 = load_frag(Kc, 16, 64, 32, lane);
      s0.v = wmma_bf16(qa0, b00, zro);
      s0.v = wmma_bf16(qa1, b01, s0.v);
      s1.v = wmma_bf16(qa0, b10, zro);
      s1.v = wmma_bf16(qa1, b11, s1.v);
    }

    // online softmax; row = i + 8*half, a row's 16 columns live in one
    // accumulator slot across a 16-lane group -> shfl_xor masks 1,2,4,8.
#pragma unroll
    for (int i = 0; i < 8; ++i) {
      float x0 = s0.f[i] * scale;
      float x1 = s1.f[i] * scale;
      float mx = fmaxf(x0, x1);
      mx = fmaxf(mx, __shfl_xor(mx, 1));
      mx = fmaxf(mx, __shfl_xor(mx, 2));
      mx = fmaxf(mx, __shfl_xor(mx, 4));
      mx = fmaxf(mx, __shfl_xor(mx, 8));
      const float mn = fmaxf(mrow[i], mx);
      const float al = __expf(mrow[i] - mn);
      const float p0 = __expf(x0 - mn);
      const float p1 = __expf(x1 - mn);
      float rs = p0 + p1;
      rs += __shfl_xor(rs, 1);
      rs += __shfl_xor(rs, 2);
      rs += __shfl_xor(rs, 4);
      rs += __shfl_xor(rs, 8);
      lrow[i] = lrow[i] * al + rs;
      mrow[i] = mn;
#pragma unroll
      for (int j = 0; j < 4; ++j) o[j].f[i] *= al;
      // scatter P (C layout) into this wave's LDS strip, row-major 16x32
      const int row = i + 8 * half;
      const int col = lane & 15;
      pl[row * 32 + col]      = f2bf(p0);
      pl[row * 32 + col + 16] = f2bf(p1);
    }

    // wave-synchronous LDS handoff: C-layout -> A-layout (DScnt only)
    asm volatile("s_wait_dscnt 0" ::: "memory");
    v16bf pa  = load_frag(pl, 0, 32, 0, lane);

    // O(16x64) += P(16x32) @ V(32x64); hoist all V-frag loads, then 4 WMMAs
    v16bf vb0 = load_frag(Vc,  0, 32, 0, lane);
    v16bf vb1 = load_frag(Vc, 16, 32, 0, lane);
    v16bf vb2 = load_frag(Vc, 32, 32, 0, lane);
    v16bf vb3 = load_frag(Vc, 48, 32, 0, lane);
    o[0].v = wmma_bf16(pa, vb0, o[0].v);
    o[1].v = wmma_bf16(pa, vb1, o[1].v);
    o[2].v = wmma_bf16(pa, vb2, o[2].v);
    o[3].v = wmma_bf16(pa, vb3, o[3].v);
    cur ^= 1;
  }

  // normalize and store context [b, s, h*64 + d] (bf16)
  const int b = bh >> 4, h = bh & 15;
#pragma unroll
  for (int j = 0; j < 4; ++j) {
#pragma unroll
    for (int i = 0; i < 8; ++i) {
      const int m = i + 8 * half;
      const int s = qt * 16 + m;
      const int d = h * HEAD_DIM + j * 16 + (lane & 15);
      Ctx[(size_t)(b * SEQ + s) * D_MODEL + d] = f2bf(o[j].f[i] / lrow[i]);
    }
  }
}

// ---------------------------------------------------------------- out proj
// out[t,n] = Ctx[t,:] . Wo[n,:] + bo[n]  (f32 output). Same cooperative
// LDS-staged weight-tile scheme as the QKV GEMM.
__global__ void __launch_bounds__(256)
oproj_kernel(const unsigned short* __restrict__ Ctx,
             const unsigned short* __restrict__ Wob,
             const float* __restrict__ bo,
             float* __restrict__ Out) {
  __shared__ unsigned short bbuf[2][64 * 32];

  const int tid  = threadIdx.x;
  const int lane = tid & 31;
  const int warp = tid >> 5;
  const int nt   = blockIdx.x % 16;
  const int mt   = (blockIdx.x / 16) * 8 + warp;
  const int n0   = nt * 64;

  const int brow = tid >> 2;
  const int bcol = (tid & 3) * 8;

  const v8f zro = {0.f, 0.f, 0.f, 0.f, 0.f, 0.f, 0.f, 0.f};
  FragC acc[4];
#pragma unroll
  for (int j = 0; j < 4; ++j) acc[j].v = zro;

  async_b128(Wob + (size_t)(n0 + brow) * D_MODEL + bcol, &bbuf[0][brow * 32 + bcol]);

  int cur = 0;
  for (int k0 = 0; k0 < D_MODEL; k0 += 32) {
    wait_async0();
    __syncthreads();
    if (k0 + 32 < D_MODEL)
      async_b128(Wob + (size_t)(n0 + brow) * D_MODEL + (k0 + 32) + bcol,
                 &bbuf[cur ^ 1][brow * 32 + bcol]);

    v16bf a  = load_frag(Ctx, mt * 16, D_MODEL, k0, lane);
    v16bf b0 = load_frag(bbuf[cur],  0, 32, 0, lane);
    v16bf b1 = load_frag(bbuf[cur], 16, 32, 0, lane);
    v16bf b2 = load_frag(bbuf[cur], 32, 32, 0, lane);
    v16bf b3 = load_frag(bbuf[cur], 48, 32, 0, lane);
    acc[0].v = wmma_bf16(a, b0, acc[0].v);
    acc[1].v = wmma_bf16(a, b1, acc[1].v);
    acc[2].v = wmma_bf16(a, b2, acc[2].v);
    acc[3].v = wmma_bf16(a, b3, acc[3].v);
    cur ^= 1;
  }

  const int half = lane >> 4;
#pragma unroll
  for (int j = 0; j < 4; ++j) {
    const int n = n0 + j * 16 + (lane & 15);
    const float bv = bo[n];
#pragma unroll
    for (int i = 0; i < 8; ++i) {
      const int m = mt * 16 + i + 8 * half;
      Out[(size_t)m * D_MODEL + n] = acc[j].f[i] + bv;
    }
  }
}

// ---------------------------------------------------------------- launcher
extern "C" void kernel_launch(void* const* d_in, const int* in_sizes, int n_in,
                              void* d_out, int out_size, void* d_ws, size_t ws_size,
                              hipStream_t stream) {
  const float* X    = (const float*)d_in[0];   // [B,S,D]
  const float* Wqkv = (const float*)d_in[1];   // [3D,D]
  const float* bqkv = (const float*)d_in[2];   // [3D]
  const float* Wo   = (const float*)d_in[3];   // [D,D]
  const float* bo   = (const float*)d_in[4];   // [D]
  float* Out = (float*)d_out;

  char* ws = (char*)d_ws;
  unsigned short* Xb    = (unsigned short*)(ws);                       // 8 MB
  unsigned short* Wqkvb = (unsigned short*)(ws + ( 8ull << 20));       // 6 MB
  unsigned short* Wob   = (unsigned short*)(ws + (14ull << 20));       // 2 MB
  unsigned short* Qb    = (unsigned short*)(ws + (16ull << 20));       // 8 MB
  unsigned short* Kb    = (unsigned short*)(ws + (24ull << 20));       // 8 MB
  unsigned short* Vtb   = (unsigned short*)(ws + (32ull << 20));       // 8 MB
  unsigned short* Ctx   = (unsigned short*)(ws + (40ull << 20));       // 8 MB

  cvt_kernel<<<(NTOK * D_MODEL + 255) / 256, 256, 0, stream>>>(X, Xb, NTOK * D_MODEL);
  cvt_kernel<<<(3 * D_MODEL * D_MODEL + 255) / 256, 256, 0, stream>>>(Wqkv, Wqkvb, 3 * D_MODEL * D_MODEL);
  cvt_kernel<<<(D_MODEL * D_MODEL + 255) / 256, 256, 0, stream>>>(Wo, Wob, D_MODEL * D_MODEL);

  // 32 m-groups (8 m-tiles each) x 48 n-strips
  qkv_gemm_kernel<<<1536, 256, 0, stream>>>(Xb, Wqkvb, bqkv, Qb, Kb, Vtb);
  // 32 (b,h) x 16 query groups (8 q-tiles each)
  attn_kernel<<<512, 256, 0, stream>>>(Qb, Kb, Vtb, Ctx);
  // 32 m-groups x 16 n-strips
  oproj_kernel<<<512, 256, 0, stream>>>(Ctx, Wob, bo, Out);
}
